// CGCNNConvSimple_2156073582918
// MI455X (gfx1250) — matrix operand order repaired and checked by reading, not compiled, not run on hardware
//
#include <hip/hip_runtime.h>
#include <hip/hip_bf16.h>

// MI455X / gfx1250 CGCNN conv.
// BF16 WMMA GEMM (v_wmma_f32_16x16x32_bf16), fp32 accumulate.
// Pipeline: k0 init -> pass1 (GEMM stats) -> k2 coef -> pass2 (GEMM + act + scatter)
//           -> k4 node stats -> k5 final softplus.

typedef __attribute__((ext_vector_type(16))) __bf16 v16bf;
typedef __attribute__((ext_vector_type(8)))  float  v8f;

#define BN_EPS 1e-5f

__device__ __forceinline__ unsigned short f32_to_bf16(float x) {
  unsigned u = __float_as_uint(x);
  if ((u & 0x7f800000u) == 0x7f800000u && (u & 0x007fffffu)) {
    return (unsigned short)((u >> 16) | 0x0040u);  // quiet NaN
  }
  unsigned r = u + 0x7fffu + ((u >> 16) & 1u);     // round-to-nearest-even
  return (unsigned short)(r >> 16);
}
__device__ __forceinline__ unsigned pack_bf16x2(float a, float b) {
  return (unsigned)f32_to_bf16(a) | ((unsigned)f32_to_bf16(b) << 16);
}
__device__ __forceinline__ float softplus_f(float x) {
  return (x > 20.f) ? x : log1pf(__expf(x));
}

// ---------------------------------------------------------------------------
// k0: zero output/stat buffers, convert weights fp32 -> bf16 in B-fragment
// layout: wbuf[((m*5+kc)*4+nt)*512 + lane*16 + j]
//   N = nt*16 + (lane&15), K = kc*32 + (lane>=16 ? 16 : 0) + j
// ---------------------------------------------------------------------------
__global__ void k0_init(const float* __restrict__ Wint, const float* __restrict__ Wupd,
                        float* __restrict__ out, float* __restrict__ gstats,
                        float* __restrict__ nstats, unsigned short* __restrict__ wbuf,
                        long outN) {
  long stride = (long)gridDim.x * blockDim.x;
  long gid = (long)blockIdx.x * blockDim.x + threadIdx.x;
  for (long i = gid; i < outN; i += stride) out[i] = 0.f;
  for (long i = gid; i < 20480; i += stride) {
    int j    = (int)(i & 15);
    int lane = (int)((i >> 4) & 31);
    int nt   = (int)((i >> 9) & 3);
    int kc   = (int)((i >> 11) % 5);
    int m    = (int)(i / 10240);
    int N = nt * 16 + (lane & 15);
    int K = kc * 32 + ((lane >= 16) ? 16 : 0) + j;
    const float* W = m ? Wupd : Wint;
    wbuf[i] = f32_to_bf16(W[K * 64 + N]);
  }
  for (long i = gid; i < 256; i += stride) gstats[i] = 0.f;
  for (long i = gid; i < 128; i += stride) nstats[i] = 0.f;
}

// ---------------------------------------------------------------------------
// Edge GEMM pass. 256 threads = 8 waves; each wave owns a 16-edge tile.
// STATS=true : accumulate per-feature sum / sumsq of z@W for both matrices.
// STATS=false: apply folded BN affine + sigmoid/softplus, scatter to agg.
// ---------------------------------------------------------------------------
template <bool STATS>
__global__ __launch_bounds__(256) void edge_pass(
    const float* __restrict__ node, const float* __restrict__ efeat,
    const int* __restrict__ srcIdx, const int* __restrict__ dstIdx,
    const unsigned short* __restrict__ wbuf,
    const float* __restrict__ coef,   // [a_int|c_int|a_upd|c_upd] x64 (pass 2)
    float* __restrict__ gstats,       // [sum_i|sq_i|sum_u|sq_u] x64 (pass 1)
    float* __restrict__ agg,          // [N,64] accumulator == d_out (pass 2)
    int nEdges, int nTiles) {
  __shared__ unsigned zbuf[8][16][84];  // 16 edges x 160 bf16 (84-uint padded rows)
  __shared__ float lstat[256];

  const int tid = threadIdx.x;
  const int lane = tid & 31;
  const int w = tid >> 5;
  const int tile = blockIdx.x * 8 + w;
  const bool active = tile < nTiles;       // wave-uniform

  if (STATS) lstat[tid] = 0.f;

  // ---- stage z tile: 16 edges x 80 float2 -> bf16x2 in LDS ----
  if (active) {
    int e0 = tile * 16;
    #pragma unroll 4
    for (int it = 0; it < 40; ++it) {
      int i = it * 32 + lane;        // 0..1279
      int e = i / 80;
      int p = i - e * 80;            // feature pair index
      int edge = e0 + e;
      float2 v = {0.f, 0.f};
      if (edge < nEdges) {
        const float* sp;
        if (p < 32)      sp = node  + (long)srcIdx[edge] * 64 + 2 * p;
        else if (p < 64) sp = node  + (long)dstIdx[edge] * 64 + 2 * (p - 32);
        else             sp = efeat + (long)edge * 32 + 2 * (p - 64);
        v = *(const float2*)sp;
      }
      zbuf[w][e][p] = pack_bf16x2(v.x, v.y);
    }
  }
  __syncthreads();

  union AF { v16bf v; unsigned u[8]; };
  union CF { v8f v; float f[8]; };
  CF ai[4], au[4];
  #pragma unroll
  for (int nt = 0; nt < 4; ++nt)
    #pragma unroll
    for (int r = 0; r < 8; ++r) { ai[nt].f[r] = 0.f; au[nt].f[r] = 0.f; }

  if (active) {
    const int M = lane & 15;
    const int hi4 = (lane >> 4) << 3;  // uint-pair offset for upper half-wave
    #pragma unroll
    for (int kc = 0; kc < 5; ++kc) {
      AF a;
      #pragma unroll
      for (int v = 0; v < 8; ++v) {
        int q = v + hi4;   // contiguous k = 16h + s (matches wbuf B layout)
        a.u[v] = zbuf[w][M][kc * 16 + q];
      }
      #pragma unroll
      for (int nt = 0; nt < 4; ++nt) {
        v16bf bi = *(const v16bf*)(wbuf + ((kc * 4 + nt) * 512 + lane * 16));
        v16bf bu = *(const v16bf*)(wbuf + ((20 + kc * 4 + nt) * 512 + lane * 16));
        ai[nt].v = __builtin_amdgcn_wmma_f32_16x16x32_bf16(
            false, a.v, false, bi, (short)0, ai[nt].v, false, false);
        au[nt].v = __builtin_amdgcn_wmma_f32_16x16x32_bf16(
            false, a.v, false, bu, (short)0, au[nt].v, false, false);
      }
    }
  }

  if (STATS) {
    if (active) {
      #pragma unroll
      for (int nt = 0; nt < 4; ++nt) {
        float si = 0.f, qi = 0.f, su = 0.f, qu = 0.f;
        #pragma unroll
        for (int r = 0; r < 8; ++r) {
          float yi = ai[nt].f[r], yu = au[nt].f[r];
          si += yi; qi += yi * yi; su += yu; qu += yu * yu;
        }
        si += __shfl_xor(si, 16, 32);
        qi += __shfl_xor(qi, 16, 32);
        su += __shfl_xor(su, 16, 32);
        qu += __shfl_xor(qu, 16, 32);
        if (lane < 16) {
          int f = nt * 16 + lane;
          atomicAdd(&lstat[f],       si);
          atomicAdd(&lstat[64 + f],  qi);
          atomicAdd(&lstat[128 + f], su);
          atomicAdd(&lstat[192 + f], qu);
        }
      }
    }
    __syncthreads();
    atomicAdd(&gstats[tid], lstat[tid]);
  } else {
    if (active) {
      int e0 = tile * 16;
      float Ai[4], Ci[4], Au[4], Cu[4];
      #pragma unroll
      for (int nt = 0; nt < 4; ++nt) {
        int f = nt * 16 + (lane & 15);
        Ai[nt] = coef[f];        Ci[nt] = coef[64 + f];
        Au[nt] = coef[128 + f];  Cu[nt] = coef[192 + f];
      }
      int rbase = (lane >> 4) << 3;  // rows 0..7 or 8..15
      #pragma unroll
      for (int r = 0; r < 8; ++r) {
        int e = e0 + rbase + r;
        if (e < nEdges) {
          long drow = (long)dstIdx[e] * 64;
          #pragma unroll
          for (int nt = 0; nt < 4; ++nt) {
            int f = nt * 16 + (lane & 15);
            float yi = Ai[nt] * ai[nt].f[r] + Ci[nt];
            float yu = Au[nt] * au[nt].f[r] + Cu[nt];
            float gate = __fdividef(1.f, 1.f + __expf(-yi));
            atomicAdd(agg + drow + f, gate * softplus_f(yu));
          }
        }
      }
    }
  }
}

// ---------------------------------------------------------------------------
// k2: fold edge-BN into per-feature affine. BN(zW+b) = a*(zW) + (beta - a*mean)
// ---------------------------------------------------------------------------
__global__ void k2_coef(const float* __restrict__ gstats,
                        const float* __restrict__ g_int, const float* __restrict__ be_int,
                        const float* __restrict__ g_upd, const float* __restrict__ be_upd,
                        float* __restrict__ coef, float invE) {
  int t = threadIdx.x;  // 128
  if (t < 64) {
    float mean = gstats[t] * invE;
    float var = fmaxf(gstats[64 + t] * invE - mean * mean, 0.f);
    float a = g_int[t] * rsqrtf(var + BN_EPS);
    coef[t] = a;
    coef[64 + t] = be_int[t] - a * mean;
  } else {
    int f = t - 64;
    float mean = gstats[128 + f] * invE;
    float var = fmaxf(gstats[192 + f] * invE - mean * mean, 0.f);
    float a = g_upd[f] * rsqrtf(var + BN_EPS);
    coef[128 + f] = a;
    coef[192 + f] = be_upd[f] - a * mean;
  }
}

// k4: per-feature sum/sumsq of agg over nodes (i%64 == tid%64 by construction)
__global__ void k4_nstats(const float* __restrict__ agg, float* __restrict__ nstats,
                          long total) {
  long stride = (long)gridDim.x * blockDim.x;
  float s = 0.f, q = 0.f;
  for (long i = (long)blockIdx.x * blockDim.x + threadIdx.x; i < total; i += stride) {
    float v = agg[i]; s += v; q += v * v;
  }
  int f = threadIdx.x & 63;
  atomicAdd(&nstats[f], s);
  atomicAdd(&nstats[64 + f], q);
}

// k5: out = softplus(node + BN(agg)), in place over d_out
__global__ void k5_out(const float* __restrict__ node, const float* __restrict__ nstats,
                       const float* __restrict__ g_bn, const float* __restrict__ be_bn,
                       float* __restrict__ out, long total, float invN) {
  long i = (long)blockIdx.x * blockDim.x + threadIdx.x;
  if (i >= total) return;
  int f = (int)(i & 63);
  float mean = nstats[f] * invN;
  float var = fmaxf(nstats[64 + f] * invN - mean * mean, 0.f);
  float a = g_bn[f] * rsqrtf(var + BN_EPS);
  float c = be_bn[f] - a * mean;
  out[i] = softplus_f(node[i] + a * out[i] + c);
}

extern "C" void kernel_launch(void* const* d_in, const int* in_sizes, int n_in,
                              void* d_out, int out_size, void* d_ws, size_t ws_size,
                              hipStream_t stream) {
  const float* node  = (const float*)d_in[0];
  const float* efeat = (const float*)d_in[1];
  const int*   src   = (const int*)d_in[2];
  const int*   dst   = (const int*)d_in[3];
  const float* Wint  = (const float*)d_in[4];
  const float* g_int = (const float*)d_in[6];
  const float* be_int= (const float*)d_in[7];
  const float* Wupd  = (const float*)d_in[8];
  const float* g_upd = (const float*)d_in[10];
  const float* be_upd= (const float*)d_in[11];
  const float* g_bn  = (const float*)d_in[12];
  const float* be_bn = (const float*)d_in[13];

  int  E = in_sizes[2];              // n edges
  long totalOut = (long)out_size;    // N_NODES * 64
  long Nn = totalOut / 64;

  // workspace carve (~44 KB)
  float* gstats = (float*)d_ws;                       // 256 f
  float* coef   = gstats + 256;                       // 256 f
  float* nstats = coef + 256;                         // 128 f
  unsigned short* wbuf = (unsigned short*)(nstats + 128);  // 20480 bf16

  float* out = (float*)d_out;
  int nTiles  = (E + 15) / 16;
  int gblocks = (nTiles + 7) / 8;

  k0_init<<<2048, 256, 0, stream>>>(Wint, Wupd, out, gstats, nstats, wbuf, totalOut);
  edge_pass<true><<<gblocks, 256, 0, stream>>>(node, efeat, src, dst, wbuf,
                                               nullptr, gstats, nullptr, E, nTiles);
  k2_coef<<<1, 128, 0, stream>>>(gstats, g_int, be_int, g_upd, be_upd, coef,
                                 1.f / (float)E);
  edge_pass<false><<<gblocks, 256, 0, stream>>>(node, efeat, src, dst, wbuf,
                                                coef, nullptr, out, E, nTiles);
  k4_nstats<<<512, 256, 0, stream>>>(out, nstats, totalOut);
  k5_out<<<(int)((totalOut + 255) / 256), 256, 0, stream>>>(node, nstats, g_bn, be_bn,
                                                            out, totalOut,
                                                            1.f / (float)Nn);
}
